// LuongAttention_816043786737
// MI455X (gfx1250) — matrix operand order repaired and checked by reading, not compiled, not run on hardware
//
#include <hip/hip_runtime.h>
#include <hip/hip_bf16.h>

typedef __bf16 bf16;
typedef __attribute__((ext_vector_type(16))) __bf16 v16bf;
typedef __attribute__((ext_vector_type(8)))  __bf16 v8bf;
typedef __attribute__((ext_vector_type(8)))  float  v8f;
typedef int v4i __attribute__((vector_size(16)));   // int4, matches builtin

constexpr int NB = 8;      // batch
constexpr int SE = 2048;   // S_enc
constexpr int SD = 2048;   // S_dec
constexpr int HH = 1024;   // hidden

// --------------------------------------------------------------------------
// CDNA5 async global->LDS staging (ASYNCcnt-tracked), with safe fallback.
// Builtin signature (from hipcc diagnostic): first arg is v4i in AS(1).
// --------------------------------------------------------------------------
#if defined(__HIP_DEVICE_COMPILE__) && \
    __has_builtin(__builtin_amdgcn_global_load_async_to_lds_b128)
#define HAVE_ASYNC_LDS 1
#endif

#if defined(__HIP_DEVICE_COMPILE__) && \
    __has_builtin(__builtin_amdgcn_s_wait_asynccnt)
#define ASYNC_WAIT(n) __builtin_amdgcn_s_wait_asynccnt(n)
#elif defined(HAVE_ASYNC_LDS)
#define ASYNC_WAIT(n) asm volatile("s_wait_asynccnt %0" ::"n"(n) : "memory")
#else
#define ASYNC_WAIT(n) do { } while (0)
#endif

// Copy one 64-byte row (32 bf16) from global to LDS.
__device__ __forceinline__ void stage_row64B(const bf16* __restrict__ src,
                                             bf16* lds) {
#if defined(HAVE_ASYNC_LDS)
  typedef __attribute__((address_space(1))) v4i* gp_t;
  typedef __attribute__((address_space(3))) v4i* lp_t;
  gp_t g = (gp_t)(void*)(src);
  lp_t l = (lp_t)(void*)(lds);
  __builtin_amdgcn_global_load_async_to_lds_b128(g, l, 0, 0);
  __builtin_amdgcn_global_load_async_to_lds_b128(g, l, 16, 0);
  __builtin_amdgcn_global_load_async_to_lds_b128(g, l, 32, 0);
  __builtin_amdgcn_global_load_async_to_lds_b128(g, l, 48, 0);
#else
  const v8bf* s = (const v8bf*)src;
  v8bf* d = (v8bf*)lds;
  d[0] = s[0]; d[1] = s[1]; d[2] = s[2]; d[3] = s[3];
#endif
}

// --------------------------------------------------------------------------
// Fragment layouts for V_WMMA_F32_16X16X32_BF16 (wave32):
// A 16x32 (MxK): lane l -> M=l&15; lanes 0-15 K=[0..7]+[16..23],
//                lanes 16-31 K=[8..15]+[24..31] (two 16B chunks).
// B 32x16 (KxN) from K-contiguous storage: lane l -> N=l&15;
//                lanes 0-15 K=[0..7]+[16..23], lanes 16-31 K=[8..15]+[24..31].
// D/C: VGPR r -> M = r + 8*(l>>4), N = l&15.
// --------------------------------------------------------------------------
__device__ __forceinline__ v16bf frag_a(const bf16* __restrict__ A, int lda,
                                        int m0, int k0, int lane) {
  int m  = m0 + (lane & 15);
  int kh = (lane >> 4) << 3;
  const bf16* p = A + (long)m * lda + k0 + kh;
  union { v16bf v; v8bf h[2]; } u;
  u.h[0] = *(const v8bf*)(p);
  u.h[1] = *(const v8bf*)(p + 16);
  return u.v;
}

// --------------------------------------------------------------------------
// Block GEMM core: 128 threads = 4 waves, block tile 128x128 (wave 64x64).
// B tile (128 N-rows x 32 K) double-buffered in LDS via async loads; row
// pitch 40 bf16 = 80B keeps ds_load_b128 16B-aligned and bank-conflict-free.
// A and BT must be K-contiguous; caller pre-offsets BT to the block's N rows.
// --------------------------------------------------------------------------
constexpr int LROW = 40;  // LDS row pitch in bf16

__device__ __forceinline__ void gemm_core(const bf16* __restrict__ A, int lda,
                                          const bf16* __restrict__ BT, int ldb,
                                          int K, bf16* lb, int m0, int nw,
                                          v8f acc[4][4]) {
  const int tid  = threadIdx.x;   // 0..127
  const int lane = tid & 31;
  // Prologue: stage k-step 0 into buffer 0 (thread t copies N-row t).
  stage_row64B(BT + (long)tid * ldb, lb + tid * LROW);
  for (int k0 = 0; k0 < K; k0 += 32) {
    const int cur = (k0 >> 5) & 1;
    if (k0 + 32 < K) {
      stage_row64B(BT + (long)tid * ldb + (k0 + 32),
                   lb + (cur ^ 1) * 128 * LROW + tid * LROW);
      ASYNC_WAIT(4);   // previous tile's 4 async issues complete (in-order)
    } else {
      ASYNC_WAIT(0);
    }
    __syncthreads();
    v16bf a[4], b[4];
#pragma unroll
    for (int i = 0; i < 4; ++i) a[i] = frag_a(A, lda, m0 + 16 * i, k0, lane);
#pragma unroll
    for (int j = 0; j < 4; ++j) {
      const bf16* p = lb + cur * 128 * LROW +
                      (nw + 16 * j + (lane & 15)) * LROW + ((lane >> 4) << 3);
      union { v16bf v; v8bf h[2]; } u;
      u.h[0] = *(const v8bf*)(p);
      u.h[1] = *(const v8bf*)(p + 16);
      b[j] = u.v;
    }
#pragma unroll
    for (int i = 0; i < 4; ++i)
#pragma unroll
      for (int j = 0; j < 4; ++j)
        acc[i][j] = __builtin_amdgcn_wmma_f32_16x16x32_bf16(
            false, a[i], false, b[j], (short)0, acc[i][j], false, false);
    __syncthreads();  // all reads of `cur` done before it is restaged
  }
}

// --------------------------------------------------------------------------
// Stage 0a: f32 -> bf16 elementwise convert
// --------------------------------------------------------------------------
__global__ void cvt_f32_bf16(const float* __restrict__ in,
                             bf16* __restrict__ out, long n) {
  long i = (long)blockIdx.x * blockDim.x + threadIdx.x;
  long stride = (long)gridDim.x * blockDim.x;
  for (; i < n; i += stride) out[i] = (bf16)in[i];
}

// --------------------------------------------------------------------------
// Stage 0b: transpose-convert  out[n][k] = (bf16) in[k][n] (LDS 32x32 tile)
// --------------------------------------------------------------------------
__global__ void __launch_bounds__(256)
tconv(const float* __restrict__ in, bf16* __restrict__ out, int Krows,
      int Ncols, long inBatch, long outBatch) {
  __shared__ float tile[32][33];
  long b = blockIdx.z;
  in  += b * inBatch;
  out += b * outBatch;
  int n0 = blockIdx.x * 32;
  int k0 = blockIdx.y * 32;
  int tx = threadIdx.x, ty = threadIdx.y;
#pragma unroll
  for (int i = 0; i < 32; i += 8)
    tile[ty + i][tx] = in[(long)(k0 + ty + i) * Ncols + n0 + tx];
  __syncthreads();
#pragma unroll
  for (int i = 0; i < 32; i += 8)
    out[(long)(n0 + ty + i) * Krows + k0 + tx] = (bf16)tile[tx][ty + i];
}

// --------------------------------------------------------------------------
// Stage 1: keys = enc @ Wa + bias   (M=B*SE, N=HH, K=HH), bf16 out
// --------------------------------------------------------------------------
__global__ void __launch_bounds__(128)
gemm_keys(const bf16* __restrict__ A, const bf16* __restrict__ BT,
          const float* __restrict__ bias, bf16* __restrict__ out) {
  __shared__ __align__(16) bf16 lb[2 * 128 * LROW];
  int lane = threadIdx.x & 31;
  int w = threadIdx.x >> 5;
  int m0 = blockIdx.y * 128 + (w >> 1) * 64;
  int nb = blockIdx.x * 128;
  int nw = (w & 1) * 64;
  v8f acc[4][4] = {};
  gemm_core(A, HH, BT + (long)nb * HH, HH, HH, lb, m0, nw, acc);
  int lm = (lane >> 4) * 8, ln = lane & 15;
#pragma unroll
  for (int i = 0; i < 4; ++i)
#pragma unroll
    for (int j = 0; j < 4; ++j) {
      int n = nb + nw + 16 * j + ln;
      float bv = bias[n];
#pragma unroll
      for (int r = 0; r < 8; ++r)
        out[(long)(m0 + 16 * i + lm + r) * HH + n] = (bf16)(acc[i][j][r] + bv);
    }
}

// --------------------------------------------------------------------------
// Stage 2: score = dec @ keys^T per batch (M=SD, N=SE, K=HH), f32 out
// --------------------------------------------------------------------------
__global__ void __launch_bounds__(128)
gemm_score(const bf16* __restrict__ dec, const bf16* __restrict__ keys,
           float* __restrict__ score) {
  __shared__ __align__(16) bf16 lb[2 * 128 * LROW];
  long z = blockIdx.z;
  const bf16* A  = dec  + z * (long)SD * HH;
  const bf16* BT = keys + z * (long)SE * HH;
  float* out     = score + z * (long)SD * SE;
  int lane = threadIdx.x & 31;
  int w = threadIdx.x >> 5;
  int m0 = blockIdx.y * 128 + (w >> 1) * 64;
  int nb = blockIdx.x * 128;
  int nw = (w & 1) * 64;
  v8f acc[4][4] = {};
  gemm_core(A, HH, BT + (long)nb * HH, HH, HH, lb, m0, nw, acc);
  int lm = (lane >> 4) * 8, ln = lane & 15;
#pragma unroll
  for (int i = 0; i < 4; ++i)
#pragma unroll
    for (int j = 0; j < 4; ++j)
#pragma unroll
      for (int r = 0; r < 8; ++r)
        out[(long)(m0 + 16 * i + lm + r) * SE + (nb + nw + 16 * j + ln)] =
            acc[i][j][r];
}

// --------------------------------------------------------------------------
// Stage 3: row softmax over 2048 elems, in-place f32 + bf16 copy.
// --------------------------------------------------------------------------
__global__ void __launch_bounds__(256)
softmax_rows(float* __restrict__ p, bf16* __restrict__ q) {
  constexpr int ROW = SE, T = 256, PER = ROW / T;
  long row = blockIdx.x;
  p += row * (long)ROW;
  q += row * (long)ROW;
  int t = threadIdx.x;
  float v[PER];
  float m = -3.402823466e38f;
#pragma unroll
  for (int i = 0; i < PER; ++i) {
    v[i] = p[t + i * T];
    m = fmaxf(m, v[i]);
  }
  __shared__ float smax[8];
  __shared__ float ssum[8];
#pragma unroll
  for (int o = 16; o > 0; o >>= 1) m = fmaxf(m, __shfl_xor(m, o, 32));
  if ((t & 31) == 0) smax[t >> 5] = m;
  __syncthreads();
  m = smax[0];
#pragma unroll
  for (int wv = 1; wv < 8; ++wv) m = fmaxf(m, smax[wv]);
  float sum = 0.f;
#pragma unroll
  for (int i = 0; i < PER; ++i) {
    v[i] = __expf(v[i] - m);
    sum += v[i];
  }
#pragma unroll
  for (int o = 16; o > 0; o >>= 1) sum += __shfl_xor(sum, o, 32);
  if ((t & 31) == 0) ssum[t >> 5] = sum;
  __syncthreads();
  sum = 0.f;
#pragma unroll
  for (int wv = 0; wv < 8; ++wv) sum += ssum[wv];
  float inv = 1.0f / sum;
#pragma unroll
  for (int i = 0; i < PER; ++i) {
    float a = v[i] * inv;
    p[t + i * T] = a;
    q[t + i * T] = (bf16)a;
  }
}

// --------------------------------------------------------------------------
// Stage 4: context = align @ enc per batch (M=SD, N=HH, K=SE), f32 out
// --------------------------------------------------------------------------
__global__ void __launch_bounds__(128)
gemm_context(const bf16* __restrict__ alignbf, const bf16* __restrict__ encT,
             float* __restrict__ ctx) {
  __shared__ __align__(16) bf16 lb[2 * 128 * LROW];
  long z = blockIdx.z;
  const bf16* A  = alignbf + z * (long)SD * SE;
  const bf16* BT = encT    + z * (long)HH * SE;
  float* out     = ctx     + z * (long)SD * HH;
  int lane = threadIdx.x & 31;
  int w = threadIdx.x >> 5;
  int m0 = blockIdx.y * 128 + (w >> 1) * 64;
  int nb = blockIdx.x * 128;
  int nw = (w & 1) * 64;
  v8f acc[4][4] = {};
  gemm_core(A, SE, BT + (long)nb * SE, SE, SE, lb, m0, nw, acc);
  int lm = (lane >> 4) * 8, ln = lane & 15;
#pragma unroll
  for (int i = 0; i < 4; ++i)
#pragma unroll
    for (int j = 0; j < 4; ++j)
#pragma unroll
      for (int r = 0; r < 8; ++r)
        out[(long)(m0 + 16 * i + lm + r) * HH + (nb + nw + 16 * j + ln)] =
            acc[i][j][r];
}

// --------------------------------------------------------------------------
extern "C" void kernel_launch(void* const* d_in, const int* in_sizes, int n_in,
                              void* d_out, int out_size, void* d_ws,
                              size_t ws_size, hipStream_t stream) {
  const float* enc  = (const float*)d_in[0];  // [8,2048,1024]
  const float* dec  = (const float*)d_in[1];  // [8,2048,1024]
  const float* Wa   = (const float*)d_in[2];  // [1024,1024]
  const float* bias = (const float*)d_in[3];  // [1024]

  float* ctx    = (float*)d_out;               // [8,2048,1024]
  float* alignf = ctx + (long)NB * SD * HH;    // [8,2048,2048]

  const long nEnc = (long)NB * SE * HH;
  const long nDec = (long)NB * SD * HH;

  char* w = (char*)d_ws;
  bf16* enc_bf   = (bf16*)(w);                 // 33.5 MB
  bf16* dec_bf   = (bf16*)(w + 33554432);      // 33.5 MB
  bf16* keys_bf  = (bf16*)(w + 67108864);      // 33.5 MB
  bf16* waT      = (bf16*)(w + 100663296);     //  2.1 MB
  bf16* encT     = (bf16*)(w + 102760448);     // 33.5 MB
  bf16* align_bf = (bf16*)(w + 136314880);     // 67.1 MB (total ~204 MB)

  // Stage 0: precision convert + transposes
  cvt_f32_bf16<<<8192, 256, 0, stream>>>(enc, enc_bf, nEnc);
  cvt_f32_bf16<<<8192, 256, 0, stream>>>(dec, dec_bf, nDec);
  {
    dim3 blk(32, 8, 1);
    dim3 g1(HH / 32, HH / 32, 1);
    tconv<<<g1, blk, 0, stream>>>(Wa, waT, HH, HH, 0, 0);
    dim3 g2(HH / 32, SE / 32, NB);
    tconv<<<g2, blk, 0, stream>>>(enc, encT, SE, HH, (long)SE * HH,
                                  (long)HH * SE);
  }
  // Stage 1: keys = enc @ Wa + bias
  {
    dim3 g(HH / 128, (NB * SE) / 128, 1);
    gemm_keys<<<g, 128, 0, stream>>>(enc_bf, waT, bias, keys_bf);
  }
  // Stage 2: score = dec @ keys^T -> alignment region of d_out
  {
    dim3 g(SE / 128, SD / 128, NB);
    gemm_score<<<g, 128, 0, stream>>>(dec_bf, keys_bf, alignf);
  }
  // Stage 3: softmax rows (in place) + bf16 copy
  softmax_rows<<<NB * SD, 256, 0, stream>>>(alignf, align_bf);
  // Stage 4: context = align @ enc
  {
    dim3 g(HH / 128, SD / 128, NB);
    gemm_context<<<g, 128, 0, stream>>>(align_bf, encT, ctx);
  }
  (void)in_sizes; (void)n_in; (void)out_size; (void)ws_size;
}
